// SCAESuite_19172734009800
// MI455X (gfx1250) — matrix operand order repaired and checked by reading, not compiled, not run on hardware
//
#include <hip/hip_runtime.h>
#include <stdint.h>

// Problem constants (match the reference)
#define LN  3                 // layers
#define TT  1024              // B*S tokens
#define DDIM 768              // d_model
#define FF  2048              // n_features
#define TOPK 64

typedef __attribute__((ext_vector_type(16))) __bf16 v16bf;
typedef __attribute__((ext_vector_type(8)))  float  v8f;

union FragU { uint4 q[2]; v16bf v; };

__device__ __forceinline__ unsigned short f2bf(float f) {
  union { float f; uint32_t u; } x; x.f = f;
  uint32_t u = x.u;
  // RNE float32 -> bf16
  return (unsigned short)((u + 0x7FFFu + ((u >> 16) & 1u)) >> 16);
}

// ---------------------------------------------------------------------------
// Register-blocked WMMA GEMM: C[M,N] = A[M,K] * B[N,K]^T, bf16 in, f32 accum.
// Block = 256 threads = 8 waves arranged 2(M) x 4(N); block tile 128x128.
// Each wave owns a 64x32 macro-tile = 4x2 WMMA tiles: per 32-wide k-step it
// loads 4 A frags + 2 B frags (12x b128) and issues 8 independent
// v_wmma_f32_16x16x32_bf16 (24 B/lane per WMMA vs 64 unblocked; independent
// accumulators also cover the XDL RAW latency without NOPs).
// Requires M % 128 == 0, N % 128 == 0, K % 32 == 0 (true at every call site).
// mode 0: Cf = C (+ bias[n]);  mode 1: Cf += C;
// mode 2: Cb(bf16) = mask[m*N+n] ? C : 0   (masked virtual matrix)
// Fragment layouts per CDNA5 ISA §7.12.2 (h = lane>>4):
//   A lane L: M = L&15, a[0..7] = K(8h..8h+7), a[8..15] = K(16+8h..16+8h+7)
//   B lane L: N = L&15, b[0..7] = K(8h..8h+7), b[8..15] = K(16+8h..16+8h+7)
// Both resolve to two aligned 16B loads per fragment -> global_load_b128.
// ---------------------------------------------------------------------------
__global__ __launch_bounds__(256)
void scae_gemm_bf16_nt(const unsigned short* __restrict__ A, int lda,
                       const unsigned short* __restrict__ B, int ldb,
                       int M, int N, int Kd, int mode,
                       float* __restrict__ Cf, unsigned short* __restrict__ Cb,
                       const unsigned char* __restrict__ mask,
                       const float* __restrict__ bias) {
  const int lane = threadIdx.x & 31;
  const int wave = threadIdx.x >> 5;
  const int wm   = wave >> 2;           // 0..1
  const int wn   = wave & 3;            // 0..3
  const int tileM = blockIdx.y * 128 + wm * 64;
  const int tileN = blockIdx.x * 128 + wn * 32;
  const int h   = lane >> 4;
  const int l15 = lane & 15;

  const unsigned short* Arow[4];
  const unsigned short* Brow[2];
#pragma unroll
  for (int mi = 0; mi < 4; ++mi)
    Arow[mi] = A + (size_t)(tileM + mi * 16 + l15) * lda;
#pragma unroll
  for (int ni = 0; ni < 2; ++ni)
    Brow[ni] = B + (size_t)(tileN + ni * 16 + l15) * ldb;

  v8f acc[4][2];
#pragma unroll
  for (int mi = 0; mi < 4; ++mi)
#pragma unroll
    for (int ni = 0; ni < 2; ++ni)
      acc[mi][ni] = (v8f){0.f, 0.f, 0.f, 0.f, 0.f, 0.f, 0.f, 0.f};

  for (int k = 0; k < Kd; k += 32) {
    FragU a[4], b[2];
#pragma unroll
    for (int mi = 0; mi < 4; ++mi) {
      a[mi].q[0] = *(const uint4*)(Arow[mi] + k + 8 * h);
      a[mi].q[1] = *(const uint4*)(Arow[mi] + k + 16 + 8 * h);
    }
#pragma unroll
    for (int ni = 0; ni < 2; ++ni) {
      b[ni].q[0] = *(const uint4*)(Brow[ni] + k + 8 * h);
      b[ni].q[1] = *(const uint4*)(Brow[ni] + k + 16 + 8 * h);
    }
#pragma unroll
    for (int mi = 0; mi < 4; ++mi)
#pragma unroll
      for (int ni = 0; ni < 2; ++ni)
        acc[mi][ni] = __builtin_amdgcn_wmma_f32_16x16x32_bf16(
            false, a[mi].v, false, b[ni].v, (short)0, acc[mi][ni], false, false);
  }

#pragma unroll
  for (int mi = 0; mi < 4; ++mi) {
#pragma unroll
    for (int ni = 0; ni < 2; ++ni) {
      const int ncol = tileN + ni * 16 + l15;
#pragma unroll
      for (int r = 0; r < 8; ++r) {
        const int m = tileM + mi * 16 + r + 8 * h;
        const size_t idx = (size_t)m * N + ncol;
        const float v = acc[mi][ni][r];
        if (mode == 0)      Cf[idx] = v + (bias ? bias[ncol] : 0.0f);
        else if (mode == 1) Cf[idx] += v;
        else                Cb[idx] = mask[idx] ? f2bf(v) : (unsigned short)0;
      }
    }
  }
}

// ---------------------------------------------------------------------------
// Small helper kernels (negligible FLOPs vs the GEMMs)
// ---------------------------------------------------------------------------
__global__ void cvt_f32_to_bf16(const float* __restrict__ in,
                                unsigned short* __restrict__ out, int n) {
  int i = blockIdx.x * blockDim.x + threadIdx.x;
  if (i < n) out[i] = f2bf(in[i]);
}

// W_dec [6][D][F] f32  ->  W_decT [6][F][D] bf16 (output-indexed, coalesced writes)
__global__ void transpose_wdec_bf16(const float* __restrict__ wdec,
                                    unsigned short* __restrict__ out) {
  int i = blockIdx.x * blockDim.x + threadIdx.x;
  const int per = FF * DDIM;
  if (i >= 6 * per) return;
  int a = i / per, rem = i % per;
  int f = rem / DDIM, dd = rem % DDIM;
  out[i] = f2bf(wdec[(size_t)a * per + (size_t)dd * FF + f]);
}

// xc = (x - b_dec_up) -> bf16, x is [T,D]
__global__ void center_to_bf16(const float* __restrict__ x,
                               const float* __restrict__ bdec,
                               unsigned short* __restrict__ out, int n) {
  int i = blockIdx.x * blockDim.x + threadIdx.x;
  if (i < n) out[i] = f2bf(x[i] - bdec[i % DDIM]);
}

// sumbd[dd] = sum over up_list(d) of b_dec[up][dd]
__global__ void sum_bdec_kernel(const float* __restrict__ b_dec, int d,
                                float* __restrict__ out) {
  int dd = blockIdx.x * blockDim.x + threadIdx.x;
  if (dd >= DDIM) return;
  float s = 0.f;
  for (int u = 0; u <= d; ++u)      s += b_dec[u * DDIM + dd];    // attn_0..attn_d
  for (int u = LN; u < LN + d; ++u) s += b_dec[u * DDIM + dd];    // mlp_0..mlp_{d-1}
  out[dd] = s;
}

// v1[g] = W_enc_down[g,:] . sumbd   (sum of all pair b_dec contributions)
// v2[g] = b_enc_down[g] - W_enc_down[g,:] . b_dec_down
__global__ void layer_vec_kernel(const float* __restrict__ W_enc,
                                 const float* __restrict__ b_enc,
                                 const float* __restrict__ b_dec,
                                 const float* __restrict__ sumbd, int down,
                                 float* __restrict__ v1, float* __restrict__ v2) {
  int g = blockIdx.x * blockDim.x + threadIdx.x;
  if (g >= FF) return;
  const float* w  = W_enc + ((size_t)down * FF + g) * DDIM;
  const float* bd = b_dec + (size_t)down * DDIM;
  float a1 = 0.f, a2 = 0.f;
  for (int dd = 0; dd < DDIM; ++dd) { float wv = w[dd]; a1 += wv * sumbd[dd]; a2 += wv * bd[dd]; }
  v1[g] = a1;
  v2[g] = b_enc[(size_t)down * FF + g] - a2;
}

// Exact top-K per row with transform v = (x + v1)*s^-1 + v2 (optional relu),
// scatter the K winners into a zeroed bf16 row (input to the next GEMM).
__global__ __launch_bounds__(256)
void topk_scatter_kernel(const float* __restrict__ x,
                         const float* __restrict__ v1,
                         const float* __restrict__ v2,
                         const float* __restrict__ srow, int relu,
                         unsigned short* __restrict__ out) {
  __shared__ float vals[FF];
  __shared__ float rv[256];
  __shared__ int   ri[256];
  __shared__ float sinv;
  const int row = blockIdx.x, tid = threadIdx.x;
  if (tid == 0) sinv = srow ? (1.0f / srow[row]) : 1.0f;
  __syncthreads();
  const float* xr = x + (size_t)row * FF;
  unsigned short* orow = out + (size_t)row * FF;
  for (int j = tid; j < FF; j += 256) {
    float v = xr[j];
    if (v1) v += v1[j];
    v *= sinv;
    if (v2) v += v2[j];
    if (relu && v < 0.f) v = 0.f;
    vals[j] = v;
    orow[j] = 0;                      // zero the sparse output row
  }
  __syncthreads();
  for (int it = 0; it < TOPK; ++it) {
    float best = -__builtin_inff(); int bi = -1;
    for (int j = tid; j < FF; j += 256) {
      float v = vals[j];
      if (v > best || (v == best && j < bi)) { best = v; bi = j; }
    }
    rv[tid] = best; ri[tid] = bi;
    __syncthreads();
    for (int s = 128; s > 0; s >>= 1) {
      if (tid < s) {
        float ov = rv[tid + s]; int oi = ri[tid + s];
        if (ov > rv[tid] || (ov == rv[tid] && oi < ri[tid])) { rv[tid] = ov; ri[tid] = oi; }
      }
      __syncthreads();
    }
    if (tid == 0) { int w = ri[0]; orow[w] = f2bf(rv[0]); vals[w] = -__builtin_inff(); }
    __syncthreads();
  }
}

// ---------------------------------------------------------------------------
// Launcher
// ---------------------------------------------------------------------------
extern "C" void kernel_launch(void* const* d_in, const int* in_sizes, int n_in,
                              void* d_out, int out_size, void* d_ws, size_t ws_size,
                              hipStream_t stream) {
  (void)in_sizes; (void)n_in; (void)out_size; (void)ws_size;
  const float* resid = (const float*)d_in[0];                 // [T,D]
  const float* ln1   = (const float*)d_in[1];                 // [3,T,D]
  const float* ln2   = (const float*)d_in[2];                 // [3,T,D]
  const float* ln2s  = (const float*)d_in[3];                 // [3,T]
  const float* W_enc = (const float*)d_in[4];                 // [6,F,D]
  const float* b_enc = (const float*)d_in[5];                 // [6,F]
  const float* W_dec = (const float*)d_in[6];                 // [6,D,F]
  const float* b_dec = (const float*)d_in[7];                 // [6,D]
  const unsigned char* masks = (const unsigned char*)d_in[8]; // [9,F,F] bool(u8)
  float* out = (float*)d_out;                                 // [3,T,D]

  // Workspace carve-up (256B aligned)
  char* ws = (char*)d_ws;
  size_t off = 0;
  auto take = [&](size_t bytes) { char* p = ws + off; off = (off + bytes + 255) & ~(size_t)255; return p; };
  unsigned short* wencB = (unsigned short*)take((size_t)6 * FF * DDIM * 2);   // W_enc bf16 [6,F,D]
  unsigned short* wdecB = (unsigned short*)take((size_t)6 * DDIM * FF * 2);   // W_dec bf16 [6,D,F]
  unsigned short* wdecT = (unsigned short*)take((size_t)6 * FF * DDIM * 2);   // W_dec^T bf16 [6,F,D]
  unsigned short* residB= (unsigned short*)take((size_t)TT * DDIM * 2);       // resid bf16
  unsigned short* xcB   = (unsigned short*)take((size_t)TT * DDIM * 2);       // centered x bf16 (reused)
  unsigned short* feats = (unsigned short*)take((size_t)5 * TT * FF * 2);     // encoded feats bf16 (ups 0..4)
  float*          preF  = (float*)take((size_t)TT * FF * 4);                  // pre / approx f32 (reused)
  unsigned short* mwB   = (unsigned short*)take((size_t)FF * FF * 2);         // masked virtual bf16 (reused)
  unsigned short* featk = (unsigned short*)take((size_t)TT * FF * 2);         // top-k feature bf16
  float*          sumbdF= (float*)take(DDIM * 4);
  float*          v1F   = (float*)take(FF * 4);
  float*          v2F   = (float*)take(FF * 4);

  const int CB = 256;
  auto blk = [&](int n) { return (n + CB - 1) / CB; };

  // 1) One-time operand conversions (re-done every call for determinism)
  cvt_f32_to_bf16<<<blk(6 * FF * DDIM), CB, 0, stream>>>(W_enc, wencB, 6 * FF * DDIM);
  cvt_f32_to_bf16<<<blk(6 * DDIM * FF), CB, 0, stream>>>(W_dec, wdecB, 6 * DDIM * FF);
  transpose_wdec_bf16<<<blk(6 * FF * DDIM), CB, 0, stream>>>(W_dec, wdecT);
  cvt_f32_to_bf16<<<blk(TT * DDIM), CB, 0, stream>>>(resid, residB, TT * DDIM);

  // 2) Encode the 5 upstream modules once: feats[up] = topk(relu((x - b_dec)W_enc^T + b_enc))
  for (int up = 0; up < 5; ++up) {
    const float* xup = (up < LN) ? (ln1 + (size_t)up * TT * DDIM)
                                 : (ln2 + (size_t)(up - LN) * TT * DDIM);
    center_to_bf16<<<blk(TT * DDIM), CB, 0, stream>>>(xup, b_dec + up * DDIM, xcB, TT * DDIM);
    scae_gemm_bf16_nt<<<dim3(FF / 128, TT / 128), CB, 0, stream>>>(
        xcB, DDIM, wencB + (size_t)up * FF * DDIM, DDIM,
        TT, FF, DDIM, /*mode=*/0, preF, nullptr, nullptr, nullptr);
    topk_scatter_kernel<<<TT, CB, 0, stream>>>(
        preF, nullptr, b_enc + (size_t)up * FF, nullptr, /*relu=*/1,
        feats + (size_t)up * TT * FF);
  }

  // 3) Per downstream mlp layer
  int pair = 0;
  for (int d = 0; d < LN; ++d) {
    const int down = LN + d;
    // approx = resid @ W_enc_down^T
    scae_gemm_bf16_nt<<<dim3(FF / 128, TT / 128), CB, 0, stream>>>(
        residB, DDIM, wencB + (size_t)down * FF * DDIM, DDIM,
        TT, FF, DDIM, /*mode=*/0, preF, nullptr, nullptr, nullptr);
    // per-column epilogue vectors
    sum_bdec_kernel<<<blk(DDIM), CB, 0, stream>>>(b_dec, d, sumbdF);
    layer_vec_kernel<<<blk(FF), CB, 0, stream>>>(W_enc, b_enc, b_dec, sumbdF, down, v1F, v2F);

    auto do_pair = [&](int up) {
      // masked virtual: mw[g,f] = mask ? W_enc_down[g,:] . W_decT_up[f,:] : 0
      scae_gemm_bf16_nt<<<dim3(FF / 128, FF / 128), CB, 0, stream>>>(
          wencB + (size_t)down * FF * DDIM, DDIM,
          wdecT + (size_t)up * FF * DDIM, DDIM,
          FF, FF, DDIM, /*mode=*/2, nullptr, mwB,
          masks + (size_t)pair * FF * FF, nullptr);
      // approx += feats_up @ mw^T
      scae_gemm_bf16_nt<<<dim3(FF / 128, TT / 128), CB, 0, stream>>>(
          feats + (size_t)up * TT * FF, FF, mwB, FF,
          TT, FF, FF, /*mode=*/1, preF, nullptr, nullptr, nullptr);
      ++pair;
    };
    for (int u = 0; u <= d; ++u)       do_pair(u);        // attn_0..attn_d
    for (int u = LN; u < LN + d; ++u)  do_pair(u);        // mlp_0..mlp_{d-1}

    // feat = topk((approx + v1)/s + v2)   (no relu)
    topk_scatter_kernel<<<TT, CB, 0, stream>>>(
        preF, v1F, v2F, ln2s + (size_t)d * TT, /*relu=*/0, featk);
    // recon = feat @ W_dec_down^T + b_dec_down -> out[d]
    scae_gemm_bf16_nt<<<dim3(DDIM / 128, TT / 128), CB, 0, stream>>>(
        featk, FF, wdecB + (size_t)down * DDIM * FF, FF,
        TT, DDIM, FF, /*mode=*/0, out + (size_t)d * TT * DDIM, nullptr, nullptr,
        b_dec + (size_t)down * DDIM);
  }
}
